// RandLANetRes_20358144983143
// MI455X (gfx1250) — compile-verified, matrix operand order checked by reading, not run on hardware
//
#include <hip/hip_runtime.h>
#include <hip/hip_bf16.h>

typedef __attribute__((ext_vector_type(16))) _Float16 v16h;
typedef __attribute__((ext_vector_type(8)))  _Float16 v8h;
typedef __attribute__((ext_vector_type(8)))  float    v8f;

#define F_CAT   128
#define TILE_E  128
#define LDS_STR 136   // 128 + 8 halfs pad -> 272B row, kills bank conflicts

// Build a 16x16x32 f16 WMMA fragment: per-lane layout is
// K = off+0..off+7 (v8h lo) and K = off+16..off+23 (v8h hi), off = 8*(lane>=16).
template <typename P>
__device__ __forceinline__ v16h frag_load(const P* p) {
    v8h lo = *(const v8h*)(p);
    v8h hi = *(const v8h*)(p + 16);
    return __builtin_shufflevector(lo, hi, 0,1,2,3,4,5,6,7,8,9,10,11,12,13,14,15);
}

__global__ void k_zero(float* __restrict__ p, size_t n) {
    size_t i = (size_t)blockIdx.x * 256 + threadIdx.x;
    if (i < n) p[i] = 0.f;
}

// Convert + transpose both 128x128 weights to f16 N-major: WT[n*128+k] = W[k*128+n]
__global__ void k_prep(const float* __restrict__ Wa, const float* __restrict__ Wg,
                       _Float16* __restrict__ WaT, _Float16* __restrict__ WgT) {
    int i = blockIdx.x * 256 + threadIdx.x;
    if (i < F_CAT * F_CAT) {
        int n = i >> 7, k = i & 127;
        WaT[i] = (_Float16)Wa[k * F_CAT + n];
        WgT[i] = (_Float16)Wg[k * F_CAT + n];
    }
}

__global__ __launch_bounds__(256)
void k_edge(const float* __restrict__ x, const float* __restrict__ pos,
            const int* __restrict__ ei,
            const float* __restrict__ Wpos, const float* __restrict__ bpos,
            const float* __restrict__ batt,
            const _Float16* __restrict__ WattT,
            float* __restrict__ aggr, int E)
{
    __shared__ _Float16 sFij[TILE_E][LDS_STR];
    __shared__ float sWpos[10 * 64];
    __shared__ float sBpos[64];
    __shared__ float sBatt[F_CAT];
    __shared__ int   sSrc[TILE_E];
    __shared__ int   sTgt[TILE_E];

    const int t = threadIdx.x;
    const int base = blockIdx.x * TILE_E;

    for (int i = t; i < 640; i += 256) sWpos[i] = Wpos[i];
    if (t < 64)               sBpos[t] = bpos[t];
    if (t >= 64 && t < 192)   sBatt[t - 64] = batt[t - 64];
    if (t < TILE_E) {
        int e = base + t;
        sSrc[t] = (e < E) ? ei[e]     : 0;
        sTgt[t] = (e < E) ? ei[E + e] : 0;
    }
    __syncthreads();

    // ---- Phase A: build fij = [x_j | relu(rel @ Wpos + b)] in LDS (f16) ----
    for (int idx = t; idx < TILE_E * 64; idx += 256) {
        int eb = idx >> 6, c = idx & 63;
        int e = base + eb;
        float v = 0.f;
        if (e < E) v = x[(size_t)sSrc[eb] * 64 + c];
        sFij[eb][c] = (_Float16)v;
    }
    {
        int eb = t >> 1;
        int c0 = (t & 1) * 32;
        int e = base + eb;
        if (e < E) {
            int j = sSrc[eb], ii = sTgt[eb];
            float pix = pos[3*ii+0], piy = pos[3*ii+1], piz = pos[3*ii+2];
            float pjx = pos[3*j+0],  pjy = pos[3*j+1],  pjz = pos[3*j+2];
            float vx = pix - pjx, vy = piy - pjy, vz = piz - pjz;
            float d = sqrtf(vx*vx + vy*vy + vz*vz);
            float rel[10] = {pix,piy,piz,pjx,pjy,pjz,vx,vy,vz,d};
            for (int c = 0; c < 32; ++c) {
                int cc = c0 + c;
                float acc = sBpos[cc];
#pragma unroll
                for (int k = 0; k < 10; ++k) acc += rel[k] * sWpos[k*64 + cc];
                sFij[eb][64 + cc] = (_Float16)fmaxf(acc, 0.f);
            }
        } else {
            for (int c = 0; c < 32; ++c) sFij[eb][64 + c0 + c] = (_Float16)0.f;
        }
    }
    __syncthreads();

    // ---- Phase B: g = relu(fij @ Watt + b) via WMMA; accumulators in regs ----
    const int w  = t >> 5, l = t & 31;
    const int lm = l & 15, hi = l >> 4, off = hi * 8;

    v16h a[4];
#pragma unroll
    for (int kk = 0; kk < 4; ++kk)
        a[kk] = frag_load(&sFij[w*16 + lm][kk*32 + off]);

    v8f g[8];
#pragma unroll
    for (int nt = 0; nt < 8; ++nt) {
        int n = nt*16 + lm;
        v8f acc = {};
#pragma unroll
        for (int kk = 0; kk < 4; ++kk) {
            v16h b = frag_load(&WattT[(size_t)n * F_CAT + kk*32 + off]);
            acc = __builtin_amdgcn_wmma_f32_16x16x32_f16(
                    false, a[kk], false, b, (short)0, acc, false, false);
        }
        float bias = sBatt[n];
#pragma unroll
        for (int r = 0; r < 8; ++r) acc[r] = fmaxf(acc[r] + bias, 0.f);
        g[nt] = acc;
    }

    // ---- Phase C: per-row channel softmax in registers + atomic scatter ----
    // D layout: element r of v8f is row m = r + 8*hi, column n = nt*16 + lm.
    // A row's 128 values live in one v8f slot across 8 nt tiles x 16 lanes of
    // one half-wave, so xor-shuffles with masks 1,2,4,8 reduce the row.
#pragma unroll
    for (int r = 0; r < 8; ++r) {
        int m  = r + hi * 8;
        int eb = w * 16 + m;
        float mx = g[0][r];
#pragma unroll
        for (int nt = 1; nt < 8; ++nt) mx = fmaxf(mx, g[nt][r]);
        mx = fmaxf(mx, __shfl_xor(mx, 1, 32));
        mx = fmaxf(mx, __shfl_xor(mx, 2, 32));
        mx = fmaxf(mx, __shfl_xor(mx, 4, 32));
        mx = fmaxf(mx, __shfl_xor(mx, 8, 32));
        float ex[8], sum = 0.f;
#pragma unroll
        for (int nt = 0; nt < 8; ++nt) { ex[nt] = __expf(g[nt][r] - mx); sum += ex[nt]; }
        sum += __shfl_xor(sum, 1, 32);
        sum += __shfl_xor(sum, 2, 32);
        sum += __shfl_xor(sum, 4, 32);
        sum += __shfl_xor(sum, 8, 32);
        float inv = 1.f / sum;
        int e = base + eb;
        if (e < E) {
            float* dst = aggr + (size_t)sTgt[eb] * F_CAT;
#pragma unroll
            for (int nt = 0; nt < 8; ++nt) {
                int n = nt*16 + lm;
                float f = (float)sFij[eb][n];
                atomicAdd(dst + n, ex[nt] * inv * f);
            }
        }
    }
}

__global__ __launch_bounds__(256)
void k_out(const float* __restrict__ aggr, const _Float16* __restrict__ WT,
           const float* __restrict__ bias, float* __restrict__ out, int N)
{
    __shared__ _Float16 sA[TILE_E][LDS_STR];
    __shared__ float sBias[F_CAT];
    const int t = threadIdx.x;
    const int base = blockIdx.x * TILE_E;
    if (t < F_CAT) sBias[t] = bias[t];
    for (int idx = t; idx < TILE_E * F_CAT; idx += 256) {
        int rb = idx >> 7, c = idx & 127;
        int row = base + rb;
        float v = (row < N) ? aggr[(size_t)row * F_CAT + c] : 0.f;
        sA[rb][c] = (_Float16)v;
    }
    __syncthreads();

    const int w  = t >> 5, l = t & 31;
    const int lm = l & 15, hi = l >> 4, off = hi * 8;

    v16h a[4];
#pragma unroll
    for (int kk = 0; kk < 4; ++kk)
        a[kk] = frag_load(&sA[w*16 + lm][kk*32 + off]);

#pragma unroll
    for (int nt = 0; nt < 8; ++nt) {
        int n = nt*16 + lm;
        v8f acc = {};
#pragma unroll
        for (int kk = 0; kk < 4; ++kk) {
            v16h b = frag_load(&WT[(size_t)n * F_CAT + kk*32 + off]);
            acc = __builtin_amdgcn_wmma_f32_16x16x32_f16(
                    false, a[kk], false, b, (short)0, acc, false, false);
        }
        float bn = sBias[n];
#pragma unroll
        for (int r = 0; r < 8; ++r) {
            int row = base + w*16 + r + hi*8;
            if (row < N) out[(size_t)row * F_CAT + n] = fmaxf(acc[r] + bn, 0.f);
        }
    }
}

extern "C" void kernel_launch(void* const* d_in, const int* in_sizes, int n_in,
                              void* d_out, int out_size, void* d_ws, size_t ws_size,
                              hipStream_t stream) {
    const float* x     = (const float*)d_in[0];
    const float* pos   = (const float*)d_in[1];
    const int*   ei    = (const int*)d_in[2];
    const float* Wpos  = (const float*)d_in[3];
    const float* bpos  = (const float*)d_in[4];
    const float* Watt  = (const float*)d_in[5];
    const float* batt  = (const float*)d_in[6];
    const float* Wglob = (const float*)d_in[7];
    const float* bglob = (const float*)d_in[8];
    float* out = (float*)d_out;

    const int N = in_sizes[1] / 3;
    const int E = in_sizes[2] / 2;

    char* ws = (char*)d_ws;
    size_t aggrBytes = (size_t)N * F_CAT * sizeof(float);
    size_t o1 = (aggrBytes + 255) & ~(size_t)255;
    float*     aggr   = (float*)ws;
    _Float16*  WattT  = (_Float16*)(ws + o1);
    _Float16*  WglobT = (_Float16*)(ws + o1 + (size_t)F_CAT * F_CAT * sizeof(_Float16));

    size_t nAggr = (size_t)N * F_CAT;
    k_zero<<<(unsigned)((nAggr + 255) / 256), 256, 0, stream>>>(aggr, nAggr);
    k_prep<<<(F_CAT * F_CAT + 255) / 256, 256, 0, stream>>>(Watt, Wglob, WattT, WglobT);
    k_edge<<<(E + TILE_E - 1) / TILE_E, 256, 0, stream>>>(
        x, pos, ei, Wpos, bpos, batt, WattT, aggr, E);
    k_out<<<(N + TILE_E - 1) / TILE_E, 256, 0, stream>>>(aggr, WglobT, bglob, out, N);
}
